// RNNModel_3332894622042
// MI455X (gfx1250) — matrix operand order repaired and checked by reading, not compiled, not run on hardware
//
#include <hip/hip_runtime.h>
#include <stdint.h>

// ---------------------------------------------------------------------------
// RNN (2-layer, ReLU) + MLP head for MI455X (gfx1250, wave32, WMMA bf16)
// B=256, T=256, D=64, H=512, O=10
// ---------------------------------------------------------------------------

#define Bsz 256
#define Tsz 256
#define Dsz 64
#define Hsz 512

typedef __attribute__((ext_vector_type(16))) __bf16 v16bf;
typedef __attribute__((ext_vector_type(8)))  float  v8f;

struct __align__(16) U4 { uint32_t a, b, c, d; };

union Frag {
    v16bf    v;
    uint32_t u[8];
    U4       q[2];
};

__device__ __forceinline__ uint16_t f2bf(float f) {
    union { float f; uint32_t u; } x; x.f = f;
    return (uint16_t)((x.u + 0x7FFFu + ((x.u >> 16) & 1u)) >> 16);
}
__device__ __forceinline__ v8f vzero8() {
    v8f z;
#pragma unroll
    for (int i = 0; i < 8; ++i) z[i] = 0.0f;
    return z;
}
__device__ __forceinline__ v8f wmma_bf16(const Frag& a, const Frag& b, v8f c) {
    return __builtin_amdgcn_wmma_f32_16x16x32_bf16(
        /*neg_a=*/false, a.v, /*neg_b=*/false, b.v,
        /*c_mod=*/(short)0, c, /*reuse_a=*/false, /*reuse_b=*/false);
}

// ---------------------------------------------------------------------------
// Prep 1: weights fp32 -> bf16 row-major; fold biases (b_ih + b_hh).
// ---------------------------------------------------------------------------
__global__ void prep_weights_kernel(
        const float* __restrict__ wih0, const float* __restrict__ whh0,
        const float* __restrict__ wih1, const float* __restrict__ whh1,
        const float* __restrict__ bih0, const float* __restrict__ bhh0,
        const float* __restrict__ bih1, const float* __restrict__ bhh1,
        uint16_t* __restrict__ wih0b, uint16_t* __restrict__ whh0b,
        uint16_t* __restrict__ wih1b, uint16_t* __restrict__ whh1b,
        float* __restrict__ bias0, float* __restrict__ bias1) {
    int i = blockIdx.x * blockDim.x + threadIdx.x;
    if (i < Hsz * Dsz) wih0b[i] = f2bf(wih0[i]);
    if (i < Hsz * Hsz) {
        whh0b[i] = f2bf(whh0[i]);
        wih1b[i] = f2bf(wih1[i]);
        wih1b[i] = f2bf(wih1[i]);
        whh1b[i] = f2bf(whh1[i]);
    }
    if (i < Hsz) {
        bias0[i] = bih0[i] + bhh0[i];
        bias1[i] = bih1[i] + bhh1[i];
    }
}

// ---------------------------------------------------------------------------
// Prep 2: x [B][T][D] fp32  ->  xb [T*B][D] bf16  (m-major, GEMM-ready rows)
// ---------------------------------------------------------------------------
__global__ void prep_x_kernel(const float* __restrict__ x, uint16_t* __restrict__ xb) {
    int i = blockIdx.x * blockDim.x + threadIdx.x;   // i = m*64 + d
    const int m = i >> 6;
    const int d = i & 63;
    const int t = m >> 8;
    const int b = m & 255;
    xb[i] = f2bf(x[((size_t)b * Tsz + t) * Dsz + d]);
}

// ---------------------------------------------------------------------------
// Input projection GEMM (both layers):
//   xp[m][n] = sum_k ab[m][k] * wb[n][k] + bias[n]
// M = 65536, N = 512, K = KT*32. A and W are bf16 row-major.
// WG = 8 waves; wave -> 16-row M tile; WG shares a 64-column N slab.
// ---------------------------------------------------------------------------
template <int KT>
__global__ void __launch_bounds__(256)
xproj_kernel(const uint16_t* __restrict__ ab, const uint16_t* __restrict__ wb,
             const float* __restrict__ bias, float* __restrict__ xp) {
    constexpr int K = KT * 32;
    const int lane = threadIdx.x & 31;
    const int wave = threadIdx.x >> 5;
    const int half = lane >> 4;
    const int lm   = lane & 15;
    const int m_base = blockIdx.x * 128 + wave * 16;
    const int n_base = blockIdx.y * 64;

    const uint16_t* rowp = ab + (size_t)(m_base + lm) * K;

    v8f acc[4];
#pragma unroll
    for (int nt = 0; nt < 4; ++nt) acc[nt] = vzero8();

#pragma unroll(KT > 4 ? 1 : KT)
    for (int kt = 0; kt < KT; ++kt) {
        const uint16_t* ap = rowp + kt * 32 + 8 * half;
        Frag a;
        a.q[0] = *(const U4*)(ap);        // bf16 k0 .. k0+7   -> slots 0..7
        a.q[1] = *(const U4*)(ap + 16);   // bf16 k0+16..k0+23 -> slots 8..15
#pragma unroll
        for (int nt = 0; nt < 4; ++nt) {
            const int n = n_base + nt * 16 + lm;
            const uint16_t* bp = wb + (size_t)n * K + kt * 32 + 8 * half;
            Frag bf;
            bf.q[0] = *(const U4*)(bp);
            bf.q[1] = *(const U4*)(bp + 16);
            acc[nt] = wmma_bf16(a, bf, acc[nt]);
        }
    }

#pragma unroll
    for (int nt = 0; nt < 4; ++nt) {
        const int n = n_base + nt * 16 + lm;
        const float bv = bias[n];
#pragma unroll
        for (int r = 0; r < 8; ++r) {
            const int row = m_base + r + 8 * half;   // C/D layout: M = r + 8*half
            xp[(size_t)row * Hsz + n] = acc[nt][r] + bv;
        }
    }
}

// ---------------------------------------------------------------------------
// Recurrence: 16 WGs, each owns 16 batch rows for all 256 timesteps.
// Hidden state lives in LDS as *bf16* row-major (stride 520 u16 — 16B-aligned
// rows, 4-bank row skew) so A-fragments are two ds_load_b128 with ZERO VALU.
// fp32 accumulate; one relu+cvt per output per step in the epilogue.
//   h_t = relu(xp_t + h_{t-1} @ W_hh^T)
// ---------------------------------------------------------------------------
#define HSTR 520   // u16 units; 520*2 bytes/row, multiple of 16B

template <bool WRITE_SEQ, bool WRITE_LAST>
__global__ void __launch_bounds__(256)
rnn_layer_kernel(const float* __restrict__ xp, const uint16_t* __restrict__ whh,
                 uint16_t* __restrict__ hseq_out,   // bf16 [T*B][H]
                 float* __restrict__ last_out) {    // fp32 [B][H]
    __shared__ uint16_t hbuf[16 * HSTR];

    const int lane = threadIdx.x & 31;
    const int wave = threadIdx.x >> 5;
    const int half = lane >> 4;
    const int lm   = lane & 15;
    const int b0   = blockIdx.x * 16;

    for (int i = threadIdx.x; i < 16 * HSTR; i += 256) hbuf[i] = 0;  // bf16 0.0
    __syncthreads();

    for (int t = 0; t < Tsz; ++t) {
        // ---- hoisted xp loads: overlap with the WMMA K-loop ----
        float xpv[4][8];
#pragma unroll
        for (int nt = 0; nt < 4; ++nt) {
            const int n = wave * 64 + nt * 16 + lm;
#pragma unroll
            for (int r = 0; r < 8; ++r) {
                const size_t gm = (size_t)t * Bsz + b0 + r + 8 * half;
                xpv[nt][r] = xp[gm * Hsz + n];
            }
        }

        v8f acc[4];
#pragma unroll
        for (int nt = 0; nt < 4; ++nt) acc[nt] = vzero8();

        for (int kt = 0; kt < 16; ++kt) {
            const int k0 = kt * 32 + 8 * half;
            const uint16_t* hp = &hbuf[lm * HSTR + k0];
            Frag a;
            a.q[0] = *(const U4*)(hp);         // ds_load_b128
            a.q[1] = *(const U4*)(hp + 16);    // ds_load_b128

            const uint16_t* wrow = whh + kt * 32 + 8 * half;
            __builtin_prefetch(wrow + 32, 0, 3);   // next K-slab (L2-hot stream)
#pragma unroll
            for (int nt = 0; nt < 4; ++nt) {
                const int n = wave * 64 + nt * 16 + lm;
                const uint16_t* bp = wrow + (size_t)n * Hsz;
                Frag bf;
                bf.q[0] = *(const U4*)(bp);
                bf.q[1] = *(const U4*)(bp + 16);
                acc[nt] = wmma_bf16(a, bf, acc[nt]);
            }
        }

        __syncthreads();   // all reads of h_{t-1} done before overwrite

#pragma unroll
        for (int nt = 0; nt < 4; ++nt) {
            const int n = wave * 64 + nt * 16 + lm;
#pragma unroll
            for (int r = 0; r < 8; ++r) {
                const int row = r + 8 * half;
                float v = acc[nt][r] + xpv[nt][r];
                v = v > 0.0f ? v : 0.0f;
                const uint16_t vb = f2bf(v);
                hbuf[row * HSTR + n] = vb;
                if (WRITE_SEQ) {
                    const size_t gm = (size_t)t * Bsz + b0 + row;
                    hseq_out[gm * Hsz + n] = vb;
                }
            }
        }

        __syncthreads();   // h_t visible before next step's reads
    }

    if (WRITE_LAST) {
#pragma unroll
        for (int nt = 0; nt < 4; ++nt) {
            const int n = wave * 64 + nt * 16 + lm;
#pragma unroll
            for (int r = 0; r < 8; ++r) {
                const int row = r + 8 * half;
                union { uint32_t u; float f; } c;
                c.u = (uint32_t)hbuf[row * HSTR + n] << 16;   // bf16 -> f32
                last_out[(size_t)(b0 + row) * Hsz + n] = c.f;
            }
        }
    }
}

// ---------------------------------------------------------------------------
// Head: fc2(relu(fc1(last))) — 2 MFLOP, fp32 scalar, latency only.
// ---------------------------------------------------------------------------
__global__ void head_kernel(const float* __restrict__ last,
                            const float* __restrict__ fc1w, const float* __restrict__ fc1b,
                            const float* __restrict__ fc2w, const float* __restrict__ fc2b,
                            float* __restrict__ out) {
    const int b = threadIdx.x;   // 256 threads, 1 block
    const float* hr = last + (size_t)b * Hsz;
    float hmid[8];
#pragma unroll
    for (int o = 0; o < 8; ++o) {
        float s = fc1b[o];
        const float* wr = fc1w + o * Hsz;
        for (int k = 0; k < Hsz; ++k) s += hr[k] * wr[k];
        hmid[o] = s > 0.0f ? s : 0.0f;
    }
#pragma unroll
    for (int o = 0; o < 10; ++o) {
        float s = fc2b[o];
#pragma unroll
        for (int k = 0; k < 8; ++k) s += hmid[k] * fc2w[o * 8 + k];
        out[b * 10 + o] = s;
    }
}

// ---------------------------------------------------------------------------
extern "C" void kernel_launch(void* const* d_in, const int* in_sizes, int n_in,
                              void* d_out, int out_size, void* d_ws, size_t ws_size,
                              hipStream_t stream) {
    const float* x    = (const float*)d_in[0];
    const float* wih0 = (const float*)d_in[1];
    const float* whh0 = (const float*)d_in[2];
    const float* bih0 = (const float*)d_in[3];
    const float* bhh0 = (const float*)d_in[4];
    const float* wih1 = (const float*)d_in[5];
    const float* whh1 = (const float*)d_in[6];
    const float* bih1 = (const float*)d_in[7];
    const float* bhh1 = (const float*)d_in[8];
    const float* fc1w = (const float*)d_in[9];
    const float* fc1b = (const float*)d_in[10];
    const float* fc2w = (const float*)d_in[11];
    const float* fc2b = (const float*)d_in[12];

    char*  ws  = (char*)d_ws;
    size_t off = 0;
    float*    xp    = (float*)(ws + off);    off += (size_t)Tsz * Bsz * Hsz * 4; // 128 MB, reused by both layers
    uint16_t* h0b   = (uint16_t*)(ws + off); off += (size_t)Tsz * Bsz * Hsz * 2; // 64 MB bf16 layer-0 outputs
    uint16_t* xb    = (uint16_t*)(ws + off); off += (size_t)Tsz * Bsz * Dsz * 2; // 8 MB bf16 input, m-major
    uint16_t* wih0b = (uint16_t*)(ws + off); off += (size_t)Hsz * Dsz * 2;
    uint16_t* whh0b = (uint16_t*)(ws + off); off += (size_t)Hsz * Hsz * 2;
    uint16_t* wih1b = (uint16_t*)(ws + off); off += (size_t)Hsz * Hsz * 2;
    uint16_t* whh1b = (uint16_t*)(ws + off); off += (size_t)Hsz * Hsz * 2;
    float*    bias0 = (float*)(ws + off);    off += (size_t)Hsz * 4;
    float*    bias1 = (float*)(ws + off);    off += (size_t)Hsz * 4;
    float*    lastb = (float*)(ws + off);    off += (size_t)Bsz * Hsz * 4;

    // 0) weight conversion + bias fold; input conversion + relayout
    prep_weights_kernel<<<(Hsz * Hsz) / 256, 256, 0, stream>>>(
        wih0, whh0, wih1, whh1, bih0, bhh0, bih1, bhh1,
        wih0b, whh0b, wih1b, whh1b, bias0, bias1);
    prep_x_kernel<<<(Tsz * Bsz * Dsz) / 256, 256, 0, stream>>>(x, xb);

    dim3 gemm_grid(Tsz * Bsz / 128, Hsz / 64);   // 512 x 8 WGs

    // 1) layer-0 input projection (WMMA, K=64)
    xproj_kernel<2><<<gemm_grid, 256, 0, stream>>>(xb, wih0b, bias0, xp);

    // 2) layer-0 recurrence (16 WGs, bf16 hidden state in LDS, WMMA per step)
    rnn_layer_kernel<true, false><<<Bsz / 16, 256, 0, stream>>>(xp, whh0b, h0b, nullptr);

    // 3) layer-1 input projection from bf16 h0 (WMMA, K=512), overwrites xp
    xproj_kernel<16><<<gemm_grid, 256, 0, stream>>>(h0b, wih1b, bias1, xp);

    // 4) layer-1 recurrence, keeps only final hidden state
    rnn_layer_kernel<false, true><<<Bsz / 16, 256, 0, stream>>>(xp, whh1b, nullptr, lastb);

    // 5) head
    head_kernel<<<1, 256, 0, stream>>>(lastb, fc1w, fc1b, fc2w, fc2b, (float*)d_out);
}
